// ANIInteraction_5677946766114
// MI455X (gfx1250) — matrix operand order repaired and checked by reading, not compiled, not run on hardware
//
#include <hip/hip_runtime.h>
#include <hip/hip_bf16.h>
#include <stdint.h>

// ---------------------------------------------------------------------------
// ANI species-MLP inference for MI455X (gfx1250, wave32, WMMA).
// gather-by-species -> per-species f16 WMMA MLP -> scatter energies.
// ---------------------------------------------------------------------------

typedef __attribute__((ext_vector_type(16))) _Float16 v16h;
typedef __attribute__((ext_vector_type(8)))  float    v8f;
typedef __attribute__((ext_vector_type(4)))  _Float16 h4;

static constexpr int kNAtoms = 262144;
static constexpr int kAev    = 384;
static constexpr int kWaves  = 4;          // waves per block
static constexpr int kTileM  = 16;         // atoms per wave tile
static constexpr int kBufAH  = 16 * 384;   // halfs, layer input staging
static constexpr int kBufBH  = 16 * 160;   // halfs, widest hidden layer

union Frag { v16h h; uint32_t u[8]; };

// ---------------------------------------------------------------------------
// Pass 1: species histogram + index scatter
// ---------------------------------------------------------------------------
__global__ void ani_zero_counts(int* counts) {
    if (threadIdx.x < 4) counts[threadIdx.x] = 0;
}

__global__ void ani_scatter_species(const int* __restrict__ species, int n,
                                    int* __restrict__ counts,
                                    int* __restrict__ lists) {
    int i = blockIdx.x * blockDim.x + threadIdx.x;
    if (i < n) {
        int s = species[i] & 3;
        int p = atomicAdd(counts + s, 1);
        lists[s * kNAtoms + p] = i;
    }
}

// ---------------------------------------------------------------------------
// Pass 2: fp32 -> f16 weight conversion (weights live in L2 afterwards)
// ---------------------------------------------------------------------------
__global__ void ani_f32_to_f16(const float* __restrict__ src,
                               _Float16* __restrict__ dst, int n) {
    int i = blockIdx.x * blockDim.x + threadIdx.x;
    if (i < n) dst[i] = (_Float16)src[i];
}

// ---------------------------------------------------------------------------
// One Linear(+CELU) layer on a 16-atom tile via v_wmma_f32_16x16x32_f16.
//   inLds : 16 x K f16 activations (row-major per atom)
//   Wg    : N x K f16 weights in global (row-major, torch [out,in])
//   outLds: 16 x N f16 activations
// A-fragment layout (ISA 7.12.2, 16-bit A 16x32):
//   lane m=lane&15; k-halfpair j<4 -> K = hiA+2j, j>=4 -> K = 16+hiA+2(j-4),
//   hiA = (lane>>4)*8.
// B-fragment layout (32x16): lane n=lane&15; u[j] -> same K as A's u[j] (hiA-based, +16 for j>=4).
// ---------------------------------------------------------------------------
template <int K, int N, bool CELU>
__device__ __forceinline__ void ani_layer(const _Float16* __restrict__ inLds,
                                          const _Float16* __restrict__ Wg,
                                          const float* __restrict__ bias,
                                          _Float16* __restrict__ outLds,
                                          int lane) {
    const int m    = lane & 15;
    const int half = lane >> 4;
    const int hiA  = half * 8;
    const int hiB  = half * 16;

    for (int nt = 0; nt < N / 16; ++nt) {
        const int n = nt * 16 + m;      // output column this lane owns (B/C/D)
        v8f acc = {};
        #pragma unroll
        for (int kc = 0; kc < K / 32; ++kc) {
            Frag a, b;
            #pragma unroll
            for (int j = 0; j < 4; ++j) {
                const int ka = kc * 32 + hiA + 2 * j;
                a.u[j]     = *(const uint32_t*)&inLds[m * K + ka];
                a.u[j + 4] = *(const uint32_t*)&inLds[m * K + ka + 16];
            }
            #pragma unroll
            for (int j = 0; j < 8; ++j) {
                const int kb = kc * 32 + hiA + 2 * (j & 3) + (j >> 2) * 16;
                b.u[j] = *(const uint32_t*)&Wg[n * K + kb];
            }
            acc = __builtin_amdgcn_wmma_f32_16x16x32_f16(
                false, a.h, false, b.h, (short)0, acc, false, false);
        }
        const float bv = bias[n];
        #pragma unroll
        for (int r = 0; r < 8; ++r) {
            float x = acc[r] + bv;
            if (CELU)  // jax.nn.celu, alpha = 0.1
                x = x > 0.0f ? x : 0.1f * (__expf(x * 10.0f) - 1.0f);
            const int mrow = r + half * 8;   // C/D layout: VGPR r, lanes>=16 -> M+8
            outLds[mrow * N + n] = (_Float16)x;
        }
    }
}

// ---------------------------------------------------------------------------
// Pass 3: per-species MLP. One 16-atom tile per wave, 4 waves per block.
// ---------------------------------------------------------------------------
template <int D1, int D2, int D3>
__global__ __launch_bounds__(kWaves * 32)
void ani_species_kernel(const float* __restrict__ aev,
                        const int*   __restrict__ idxList,
                        const int*   __restrict__ countPtr,
                        float*       __restrict__ out,
                        const _Float16* __restrict__ W1, const float* __restrict__ b1,
                        const _Float16* __restrict__ W2, const float* __restrict__ b2,
                        const _Float16* __restrict__ W3, const float* __restrict__ b3,
                        const _Float16* __restrict__ W4, const float* __restrict__ b4) {
    extern __shared__ _Float16 smem[];
    const int lane = threadIdx.x & 31;
    const int wave = threadIdx.x >> 5;
    _Float16* bufA = smem + wave * (kBufAH + kBufBH);
    _Float16* bufB = bufA + kBufAH;

    const int count = countPtr[0];
    const int base  = (blockIdx.x * kWaves + wave) * kTileM;
    if (base >= count) return;     // wave-uniform: EXEC stays all-1s for WMMA

    // Atom index for "my" row (lanes 0..15 and 16..31 mirror each other).
    const int gpos   = base + (lane & 15);
    const int gclamp = gpos < count ? gpos : count - 1;
    const int atom_l = idxList[gclamp];

    // Gather 16 AEV rows (fp32) -> f16 staging in LDS. 3 passes of
    // 32 lanes x float4 = 128 floats cover one 384-float row.
    for (int r = 0; r < kTileM; ++r) {
        const int atom = __shfl(atom_l, r, 32);
        const float4* src = (const float4*)(aev + (size_t)atom * kAev) + lane;
        #pragma unroll
        for (int p = 0; p < 3; ++p) {
            const float4 v = src[p * 32];
            h4 hv;
            hv.x = (_Float16)v.x; hv.y = (_Float16)v.y;
            hv.z = (_Float16)v.z; hv.w = (_Float16)v.w;
            *(h4*)(bufA + r * kAev + p * 128 + lane * 4) = hv;
        }
    }

    ani_layer<kAev, D1, true>(bufA, W1, b1, bufB, lane);   // 384 -> D1, CELU
    ani_layer<D1,   D2, true>(bufB, W2, b2, bufA, lane);   // D1  -> D2, CELU
    ani_layer<D2,   D3, true>(bufA, W3, b3, bufB, lane);   // D2  -> D3, CELU

    // Final D3 -> 1 dot product: lanes <16 / >=16 split the K range.
    float acc = 0.0f;
    const int m  = lane & 15;
    const int k0 = (lane >> 4) * (D3 / 2);
    #pragma unroll
    for (int k = 0; k < D3 / 2; ++k)
        acc += (float)bufB[m * D3 + k0 + k] * (float)W4[k0 + k];
    acc += __shfl_xor(acc, 16, 32);

    if (lane < 16) {
        const int g = base + m;
        if (g < count) out[idxList[g]] = acc + b4[0];
    }
}

// ---------------------------------------------------------------------------
// Host-side launch. Input order (setup_inputs dict, leaves flattened):
//   d_in[0]=aev, d_in[1]=species, then for each species s in {H,C,N,O}:
//   d_in[2+8s+2l] = W_l (fp32 [out,in]), d_in[2+8s+2l+1] = b_l (fp32).
// Workspace layout:
//   [0,16)                      : int counts[4]
//   [16, 16+4*N*4)              : int lists[4][N]
//   [wbase, ...)                : f16 weights, species-major, layer-major
// ---------------------------------------------------------------------------
static const int kDims[4][5] = {
    {384, 160, 128, 96, 1},   // H
    {384, 144, 112, 96, 1},   // C
    {384, 128, 112, 96, 1},   // N
    {384, 128, 112, 96, 1},   // O
};

extern "C" void kernel_launch(void* const* d_in, const int* in_sizes, int n_in,
                              void* d_out, int out_size, void* d_ws, size_t ws_size,
                              hipStream_t stream) {
    (void)in_sizes; (void)n_in; (void)out_size; (void)ws_size;

    const float* aev     = (const float*)d_in[0];
    const int*   species = (const int*)d_in[1];
    float*       out     = (float*)d_out;

    char* ws      = (char*)d_ws;
    int*  counts  = (int*)ws;
    int*  lists   = (int*)(ws + 16);
    _Float16* wbase = (_Float16*)(ws + 16 + (size_t)4 * kNAtoms * 4);

    // Pass 1: species histogram + index scatter.
    ani_zero_counts<<<1, 32, 0, stream>>>(counts);
    ani_scatter_species<<<(kNAtoms + 255) / 256, 256, 0, stream>>>(
        species, kNAtoms, counts, lists);

    // Pass 2: convert weights to f16 into workspace.
    const _Float16* Wf[4][4];
    const float*    Bf[4][4];
    size_t off = 0;
    for (int s = 0; s < 4; ++s) {
        for (int l = 0; l < 4; ++l) {
            const int nW = kDims[s][l + 1] * kDims[s][l];
            const float* Wsrc = (const float*)d_in[2 + s * 8 + 2 * l];
            Bf[s][l] = (const float*)d_in[2 + s * 8 + 2 * l + 1];
            _Float16* dst = wbase + off;
            Wf[s][l] = dst;
            ani_f32_to_f16<<<(nW + 255) / 256, 256, 0, stream>>>(Wsrc, dst, nW);
            off += (size_t)nW;
        }
    }

    // Pass 3: per-species WMMA MLP. Grid sized for worst case; waves whose
    // tile exceeds the (device-side) species count exit immediately.
    const int blocks = (kNAtoms + kWaves * kTileM - 1) / (kWaves * kTileM);
    const size_t lds = (size_t)kWaves * (kBufAH + kBufBH) * sizeof(_Float16);

    ani_species_kernel<160, 128, 96><<<blocks, kWaves * 32, lds, stream>>>(
        aev, lists + 0 * kNAtoms, counts + 0, out,
        Wf[0][0], Bf[0][0], Wf[0][1], Bf[0][1], Wf[0][2], Bf[0][2], Wf[0][3], Bf[0][3]);
    ani_species_kernel<144, 112, 96><<<blocks, kWaves * 32, lds, stream>>>(
        aev, lists + 1 * kNAtoms, counts + 1, out,
        Wf[1][0], Bf[1][0], Wf[1][1], Bf[1][1], Wf[1][2], Bf[1][2], Wf[1][3], Bf[1][3]);
    ani_species_kernel<128, 112, 96><<<blocks, kWaves * 32, lds, stream>>>(
        aev, lists + 2 * kNAtoms, counts + 2, out,
        Wf[2][0], Bf[2][0], Wf[2][1], Bf[2][1], Wf[2][2], Bf[2][2], Wf[2][3], Bf[2][3]);
    ani_species_kernel<128, 112, 96><<<blocks, kWaves * 32, lds, stream>>>(
        aev, lists + 3 * kNAtoms, counts + 3, out,
        Wf[3][0], Bf[3][0], Wf[3][1], Bf[3][1], Wf[3][2], Bf[3][2], Wf[3][3], Bf[3][3]);
}
